// SequenceHead_89429809037776
// MI455X (gfx1250) — matrix operand order repaired and checked by reading, not compiled, not run on hardware
//
#include <hip/hip_runtime.h>
#include <hip/hip_bf16.h>
#include <stdint.h>

// ---------------- problem dims ----------------
#define TOK   32768      // B*N = 32*1024 tokens
#define CDIM  384
#define HDIM  1024
#define NRESC 20
#define NPAD  32         // NRES padded to 32 for the WMMA tile
#define NTABD 21

typedef __attribute__((ext_vector_type(16))) __bf16 v16bf;
typedef __attribute__((ext_vector_type(8)))  float  v8f;

struct alignas(16) Frag8 { unsigned d[8]; };

__device__ __forceinline__ v16bf frag_cast(const Frag8 &f) {
  v16bf r; __builtin_memcpy(&r, &f, sizeof(r)); return r;
}

__device__ __forceinline__ unsigned short f32_to_bf16_rne(float f) {
  unsigned u = __float_as_uint(f);
  u += 0x7FFFu + ((u >> 16) & 1u);
  return (unsigned short)(u >> 16);
}

// ---------------- weight prep: fp32 (K x N) -> bf16 transposed (N x K) ----------------
__global__ void convert_transpose_kernel(const float* __restrict__ W,
                                         unsigned short* __restrict__ WT,
                                         int Krows, int Ncols) {
  int total = Krows * Ncols;
  for (int i = blockIdx.x * blockDim.x + threadIdx.x; i < total;
       i += gridDim.x * blockDim.x) {
    int n = i / Krows;
    int k = i - n * Krows;
    WT[i] = f32_to_bf16_rne(W[(size_t)k * Ncols + n]);
  }
}

// w3 (H x 20) -> bf16 transposed + zero-padded (32 x H)
__global__ void convert_w3_kernel(const float* __restrict__ W,
                                  unsigned short* __restrict__ WT) {
  int total = NPAD * HDIM;
  for (int i = blockIdx.x * blockDim.x + threadIdx.x; i < total;
       i += gridDim.x * blockDim.x) {
    int n = i / HDIM;
    int k = i - n * HDIM;
    WT[i] = (n < NRESC) ? f32_to_bf16_rne(W[(size_t)k * NRESC + n]) : (unsigned short)0;
  }
}

// ---------------- LayerNorm + bf16 cast: one wave32 per token ----------------
__global__ void ln_cast_kernel(const float* __restrict__ act,
                               const float* __restrict__ g,
                               const float* __restrict__ b,
                               unsigned short* __restrict__ h0) {
  int wavesPerBlock = blockDim.x >> 5;
  int t = blockIdx.x * wavesPerBlock + (threadIdx.x >> 5);
  int lane = threadIdx.x & 31;
  if (t >= TOK) return;
  const float* row = act + (size_t)t * CDIM;
  float s = 0.f, ss = 0.f;
  for (int i = lane; i < CDIM; i += 32) {
    float v = row[i];
    s += v; ss += v * v;
  }
  #pragma unroll
  for (int off = 16; off > 0; off >>= 1) {
    s  += __shfl_xor(s,  off, 32);
    ss += __shfl_xor(ss, off, 32);
  }
  float mu  = s * (1.0f / CDIM);
  float var = ss * (1.0f / CDIM) - mu * mu;
  float inv = rsqrtf(var + 1e-5f);
  for (int i = lane; i < CDIM; i += 32) {
    float v = (row[i] - mu) * inv * g[i] + b[i];
    h0[(size_t)t * CDIM + i] = f32_to_bf16_rne(v);
  }
}

// ---------------- bf16 WMMA GEMM + bias + ReLU -> bf16 ----------------
// A: M x K bf16 row-major.  BT: N x K bf16 row-major (B transposed).
// Block = 8 waves; each wave computes 32(M) x 64(N); block tile = 256(M) x 64(N).
// The shared B slab (64 cols x 32 K = 4 KB) is double-buffered through LDS via
// GLOBAL_LOAD_ASYNC_TO_LDS_B128 (ASYNCcnt) in WMMA per-lane fragment layout.
__global__ void gemm_relu_bf16_kernel(const unsigned short* __restrict__ A,
                                      const unsigned short* __restrict__ BT,
                                      const float* __restrict__ bias,
                                      unsigned short* __restrict__ C,
                                      int M, int N, int K) {
  __shared__ alignas(16) unsigned short bslab[2][2048];  // 2 x 4KB

  const int tid   = threadIdx.x;
  const int lane  = tid & 31;
  const int wave  = tid >> 5;
  const int tilesN = N >> 6;
  const int blkM = blockIdx.x / tilesN;
  const int blkN = blockIdx.x - blkM * tilesN;
  const int m0 = blkM * 256 + wave * 32;
  const int n0 = blkN * 64;

  const int half = lane >> 4;           // K-half selector per ISA A/B layout
  const int nl   = lane & 15;

  // --- per-thread async-copy decomposition: 256 threads x 16B = 4KB slab ---
  // chunk(tid) -> (tile, lane-of-fragment, 16B half)
  const int ctile = tid >> 6;           // 0..3 (16-col tile)
  const int crem  = tid & 63;
  const int clane = crem >> 1;          // fragment lane 0..31
  const int chalf = crem & 1;           // which 16B half of the 32B lane chunk
  const int ccol  = n0 + ctile * 16 + (clane & 15);
  const int ckoff = 8 * (clane >> 4) + 16 * chalf;
  const unsigned short* csrc_base = BT + (size_t)ccol * K + ckoff;
  const unsigned cldsoff = (unsigned)((ctile * 512 + clane * 16 + chalf * 8) * 2);
  const unsigned ldsbase0 = (unsigned)(size_t)&bslab[0][0];
  const unsigned ldsbase1 = (unsigned)(size_t)&bslab[1][0];

  // A fragment source rows for this wave (two 16-row sub-tiles)
  const unsigned short* arow0 = A + (size_t)(m0 + nl) * K;
  const unsigned short* arow1 = arow0 + (size_t)16 * K;

  v8f acc0[4] = {};
  v8f acc1[4] = {};

  // prologue: stage k0 = 0 slab into buffer 0
  {
    const unsigned short* src = csrc_base;
    unsigned dst = ldsbase0 + cldsoff;
    asm volatile("global_load_async_to_lds_b128 %0, %1, off"
                 :: "v"(dst), "v"(src) : "memory");
  }

  int buf = 0;
  for (int k0 = 0; k0 < K; k0 += 32, buf ^= 1) {
    if (k0 + 32 < K) {
      // stage next slab into the other buffer (last read in the previous iteration,
      // protected by the end-of-compute barrier below)
      const unsigned short* src = csrc_base + (k0 + 32);
      unsigned dst = (buf ? ldsbase0 : ldsbase1) + cldsoff;
      asm volatile("global_load_async_to_lds_b128 %0, %1, off"
                   :: "v"(dst), "v"(src) : "memory");
      asm volatile("s_wait_asynccnt 0x1" ::: "memory");   // current slab done
    } else {
      asm volatile("s_wait_asynccnt 0x0" ::: "memory");
    }
    __syncthreads();   // whole block's slab chunks visible

    // A fragments (direct from global; rows are unique per wave)
    Frag8 fa0, fa1;
    *(uint4*)&fa0.d[0] = *(const uint4*)(arow0 + k0 + 8 * half);
    *(uint4*)&fa0.d[4] = *(const uint4*)(arow0 + k0 + 16 + 8 * half);
    *(uint4*)&fa1.d[0] = *(const uint4*)(arow1 + k0 + 8 * half);
    *(uint4*)&fa1.d[4] = *(const uint4*)(arow1 + k0 + 16 + 8 * half);
    __builtin_prefetch(arow0 + k0 + 32, 0, 1);
    __builtin_prefetch(arow1 + k0 + 32, 0, 1);
    v16bf a0 = frag_cast(fa0);
    v16bf a1 = frag_cast(fa1);

    #pragma unroll
    for (int tle = 0; tle < 4; ++tle) {
      const uint4* bp = (const uint4*)&bslab[buf][tle * 512 + lane * 16];
      Frag8 fb;
      *(uint4*)&fb.d[0] = bp[0];
      *(uint4*)&fb.d[4] = bp[1];
      v16bf bv = frag_cast(fb);
      acc0[tle] = __builtin_amdgcn_wmma_f32_16x16x32_bf16(
          false, a0, false, bv, (short)0, acc0[tle], false, false);
      acc1[tle] = __builtin_amdgcn_wmma_f32_16x16x32_bf16(
          false, a1, false, bv, (short)0, acc1[tle], false, false);
    }
    __syncthreads();   // everyone done reading this buffer
  }

  // C/D layout: lane&15 is column; VGPR r is row r + 8*(lane>>4)
  #pragma unroll
  for (int tle = 0; tle < 4; ++tle) {
    int col = n0 + tle * 16 + nl;
    float bz = bias[col];
    #pragma unroll
    for (int r = 0; r < 8; ++r) {
      int row0 = m0 + r + 8 * half;
      float v0 = acc0[tle][r] + bz;
      v0 = v0 > 0.f ? v0 : 0.f;
      C[(size_t)row0 * N + col] = f32_to_bf16_rne(v0);
      int row1 = row0 + 16;
      float v1 = acc1[tle][r] + bz;
      v1 = v1 > 0.f ? v1 : 0.f;
      C[(size_t)row1 * N + col] = f32_to_bf16_rne(v1);
    }
  }
}

// ---------------- logits GEMM: K=HDIM, N=20 (padded 32), f32 out, no ReLU ----------------
__global__ void gemm_logits_kernel(const unsigned short* __restrict__ A,
                                   const unsigned short* __restrict__ BT, // 32 x K
                                   const float* __restrict__ bias,        // 20
                                   float* __restrict__ C,                 // M x 20
                                   int M, int K) {
  const int lane = threadIdx.x & 31;
  const int wave = threadIdx.x >> 5;
  const int m0 = blockIdx.x * 128 + wave * 16;
  if (m0 >= M) return;
  const int half = lane >> 4;
  const int ml   = m0 + (lane & 15);
  const int nl   = lane & 15;

  v8f acc[2] = {};
  const unsigned short* arow = A + (size_t)ml * K;
  for (int k0 = 0; k0 < K; k0 += 32) {
    Frag8 fa;
    *(uint4*)&fa.d[0] = *(const uint4*)(arow + k0 + 8 * half);
    *(uint4*)&fa.d[4] = *(const uint4*)(arow + k0 + 16 + 8 * half);
    v16bf av = frag_cast(fa);
    #pragma unroll
    for (int tle = 0; tle < 2; ++tle) {
      const unsigned short* brow =
          BT + (size_t)(tle * 16 + nl) * K + k0 + 8 * half;
      Frag8 fb;
      *(uint4*)&fb.d[0] = *(const uint4*)(brow);
      *(uint4*)&fb.d[4] = *(const uint4*)(brow + 16);
      acc[tle] = __builtin_amdgcn_wmma_f32_16x16x32_bf16(
          false, av, false, frag_cast(fb), (short)0, acc[tle], false, false);
    }
  }
  #pragma unroll
  for (int tle = 0; tle < 2; ++tle) {
    int col = tle * 16 + nl;
    if (col < NRESC) {
      float bz = bias[col];
      #pragma unroll
      for (int r = 0; r < 8; ++r) {
        int row = m0 + r + 8 * half;
        C[(size_t)row * NRESC + col] = acc[tle][r] + bz;
      }
    }
  }
}

// ---------------- argmax (== argmax of softmax) + fixed-mask merge ----------------
__global__ void argmax_seq_kernel(const float* __restrict__ logits,
                                  const int* __restrict__ fixed_mask,
                                  const int* __restrict__ seq_t,
                                  float* __restrict__ seq0f,
                                  int* __restrict__ seq0i) {
  int t = blockIdx.x * blockDim.x + threadIdx.x;
  if (t >= TOK) return;
  const float* l = logits + (size_t)t * NRESC;
  int best = 0; float bv = l[0];
  #pragma unroll
  for (int i = 1; i < NRESC; ++i) {
    float v = l[i];
    if (v > bv) { bv = v; best = i; }     // first-max wins, matches jnp.argmax
  }
  int fm = fixed_mask[t];
  int s = best * (1 - fm) + seq_t[t] * fm;
  seq0i[t] = s;
  seq0f[t] = (float)s;
}

// ---------------- rigid-body geometry: one thread per token ----------------
__device__ __forceinline__ void mat_mul3(const float* a, const float* b, float* o) {
  #pragma unroll
  for (int i = 0; i < 3; ++i)
    #pragma unroll
    for (int j = 0; j < 3; ++j)
      o[i*3+j] = a[i*3+0]*b[0*3+j] + a[i*3+1]*b[1*3+j] + a[i*3+2]*b[2*3+j];
}
__device__ __forceinline__ void mat_vec3(const float* a, const float* v, float* o) {
  #pragma unroll
  for (int i = 0; i < 3; ++i)
    o[i] = a[i*3+0]*v[0] + a[i*3+1]*v[1] + a[i*3+2]*v[2];
}

__global__ void geometry_kernel(const float* __restrict__ angles,
                                const float* __restrict__ rigids,
                                const int* __restrict__ seq0,
                                const int* __restrict__ idx37,
                                const float* __restrict__ dframes,   // NTAB,8,4,4
                                const int* __restrict__ gidx,        // NTAB,14
                                const float* __restrict__ a14m,      // NTAB,14
                                const float* __restrict__ a37m,      // NTAB,37
                                const float* __restrict__ litpos,    // NTAB,14,3
                                float* __restrict__ pred,            // T,14,3
                                float* __restrict__ fin,             // T,37,3
                                float* __restrict__ m14o,            // T,14
                                float* __restrict__ m37o) {          // T,37
  int t = blockIdx.x * blockDim.x + threadIdx.x;
  if (t >= TOK) return;

  // backbone rigid: quat -> rot (+ normalize with 1e-8)
  const float* rg = rigids + (size_t)t * 7;
  float q0 = rg[0], q1 = rg[1], q2 = rg[2], q3 = rg[3];
  float inv = rsqrtf(q0*q0 + q1*q1 + q2*q2 + q3*q3 + 1e-8f);
  float w = q0*inv, x = q1*inv, y = q2*inv, z = q3*inv;
  float rb[9] = {
    1.f - 2.f*(y*y + z*z), 2.f*(x*y - w*z),       2.f*(x*z + w*y),
    2.f*(x*y + w*z),       1.f - 2.f*(x*x + z*z), 2.f*(y*z - w*x),
    2.f*(x*z - w*y),       2.f*(y*z + w*x),       1.f - 2.f*(x*x + y*y)
  };
  float tb[3] = { rg[4], rg[5], rg[6] };

  int s = seq0[t];
  const float* ang = angles + (size_t)t * 14;

  // frames: fr[g] = default_rot @ Rx(angle g); ft[g] = default_trans
  float fr[8][9], ft[8][3];
  #pragma unroll
  for (int g = 0; g < 8; ++g) {
    const float* m = dframes + ((size_t)s * 8 + g) * 16;
    float sn = (g == 0) ? 0.f : ang[(g - 1) * 2 + 0];
    float cs = (g == 0) ? 1.f : ang[(g - 1) * 2 + 1];
    #pragma unroll
    for (int i = 0; i < 3; ++i) {
      float d0 = m[i*4+0], d1 = m[i*4+1], d2 = m[i*4+2];
      fr[g][i*3+0] = d0;
      fr[g][i*3+1] =  d1 * cs + d2 * sn;
      fr[g][i*3+2] = -d1 * sn + d2 * cs;
      ft[g][i] = m[i*4+3];
    }
  }
  // chi chain composition for groups 5,6,7
  #pragma unroll
  for (int g = 5; g < 8; ++g) {
    float rt[9], tt[3];
    mat_mul3(fr[g-1], fr[g], rt);
    mat_vec3(fr[g-1], ft[g], tt);
    #pragma unroll
    for (int i = 0; i < 9; ++i) fr[g][i] = rt[i];
    #pragma unroll
    for (int i = 0; i < 3; ++i) ft[g][i] = tt[i] + ft[g-1][i];
  }
  // compose with backbone -> global frames (in place)
  #pragma unroll
  for (int g = 0; g < 8; ++g) {
    float rt[9], tt[3];
    mat_mul3(rb, fr[g], rt);
    mat_vec3(rb, ft[g], tt);
    #pragma unroll
    for (int i = 0; i < 9; ++i) fr[g][i] = rt[i];
    #pragma unroll
    for (int i = 0; i < 3; ++i) ft[g][i] = tt[i] + tb[i];
  }

  // atom14 positions
  const int*   gi  = gidx   + s * 14;
  const float* lit = litpos + (size_t)s * 14 * 3;
  const float* m14 = a14m   + s * 14;
  float pl[14][3];
  #pragma unroll
  for (int a = 0; a < 14; ++a) {
    int g = gi[a];
    float l0 = lit[a*3+0], l1 = lit[a*3+1], l2 = lit[a*3+2];
    float mk = m14[a];
    #pragma unroll
    for (int i = 0; i < 3; ++i) {
      float v = fr[g][i*3+0]*l0 + fr[g][i*3+1]*l1 + fr[g][i*3+2]*l2 + ft[g][i];
      pl[a][i] = v * mk;
      pred[((size_t)t * 14 + a) * 3 + i] = pl[a][i];
    }
    m14o[(size_t)t * 14 + a] = mk;
  }
  // atom37 gather
  const int* i37 = idx37 + (size_t)t * 37;
  for (int a = 0; a < 37; ++a) {
    int k = i37[a];
    fin[((size_t)t * 37 + a) * 3 + 0] = pl[k][0];
    fin[((size_t)t * 37 + a) * 3 + 1] = pl[k][1];
    fin[((size_t)t * 37 + a) * 3 + 2] = pl[k][2];
    m37o[(size_t)t * 37 + a] = a37m[(size_t)s * 37 + a];
  }
}

// ---------------- launcher ----------------
extern "C" void kernel_launch(void* const* d_in, const int* in_sizes, int n_in,
                              void* d_out, int out_size, void* d_ws, size_t ws_size,
                              hipStream_t stream) {
  (void)in_sizes; (void)n_in; (void)out_size; (void)ws_size;
  const float* act        = (const float*)d_in[0];
  const float* angles     = (const float*)d_in[1];
  const float* rigids     = (const float*)d_in[2];
  const int*   fixed_mask = (const int*)d_in[3];
  const int*   seq_t      = (const int*)d_in[4];
  const int*   idx37      = (const int*)d_in[5];
  const float* ln_g       = (const float*)d_in[6];
  const float* ln_b       = (const float*)d_in[7];
  const float* w1         = (const float*)d_in[8];
  const float* b1         = (const float*)d_in[9];
  const float* w2         = (const float*)d_in[10];
  const float* b2         = (const float*)d_in[11];
  const float* w3         = (const float*)d_in[12];
  const float* b3         = (const float*)d_in[13];
  const float* dframes    = (const float*)d_in[14];
  const int*   gidx       = (const int*)d_in[15];
  const float* a14m       = (const float*)d_in[16];
  const float* a37m       = (const float*)d_in[17];
  const float* litpos     = (const float*)d_in[18];

  // workspace layout (bytes)
  char* ws = (char*)d_ws;
  unsigned short* h0  = (unsigned short*)ws;                 ws += (size_t)TOK * CDIM * 2;
  unsigned short* h1  = (unsigned short*)ws;                 ws += (size_t)TOK * HDIM * 2;
  unsigned short* h2  = (unsigned short*)ws;                 ws += (size_t)TOK * HDIM * 2;
  unsigned short* w1T = (unsigned short*)ws;                 ws += (size_t)HDIM * CDIM * 2;
  unsigned short* w2T = (unsigned short*)ws;                 ws += (size_t)HDIM * HDIM * 2;
  unsigned short* w3T = (unsigned short*)ws;                 ws += (size_t)NPAD * HDIM * 2;
  int*            s0i = (int*)ws;

  // output layout (floats, tuple flattened in return order)
  float* logits = (float*)d_out;
  float* seq0f  = logits + (size_t)TOK * NRESC;
  float* pred   = seq0f  + (size_t)TOK;
  float* fin    = pred   + (size_t)TOK * 14 * 3;
  float* m14o   = fin    + (size_t)TOK * 37 * 3;
  float* m37o   = m14o   + (size_t)TOK * 14;

  // 1) weight prep
  convert_transpose_kernel<<<1536, 256, 0, stream>>>(w1, w1T, CDIM, HDIM);
  convert_transpose_kernel<<<4096, 256, 0, stream>>>(w2, w2T, HDIM, HDIM);
  convert_w3_kernel<<<128, 256, 0, stream>>>(w3, w3T);

  // 2) layernorm + bf16 cast (8 waves / block -> 8 tokens / block)
  ln_cast_kernel<<<TOK / 8, 256, 0, stream>>>(act, ln_g, ln_b, h0);

  // 3) GEMM1: (TOK x 384)@(384 x 1024) + b1, ReLU  (block tile 256x64)
  gemm_relu_bf16_kernel<<<(TOK / 256) * (HDIM / 64), 256, 0, stream>>>(
      h0, w1T, b1, h1, TOK, HDIM, CDIM);

  // 4) GEMM2: (TOK x 1024)@(1024 x 1024) + b2, ReLU
  gemm_relu_bf16_kernel<<<(TOK / 256) * (HDIM / 64), 256, 0, stream>>>(
      h1, w2T, b2, h2, TOK, HDIM, HDIM);

  // 5) logits: (TOK x 1024)@(1024 x 20) + b3
  gemm_logits_kernel<<<TOK / 128, 256, 0, stream>>>(h2, w3T, b3, logits, TOK, HDIM);

  // 6) argmax + fixed-mask merge
  argmax_seq_kernel<<<TOK / 256, 256, 0, stream>>>(logits, fixed_mask, seq_t, seq0f, s0i);

  // 7) rigid-body geometry -> pred / final / masks
  geometry_kernel<<<TOK / 256, 256, 0, stream>>>(
      angles, rigids, s0i, idx37, dframes, gidx, a14m, a37m, litpos,
      pred, fin, m14o, m37o);
}
